// cosine_16011638079614
// MI455X (gfx1250) — matrix operand order repaired and checked by reading, compiled, not run on hardware
//
#include <hip/hip_runtime.h>

#define N_ROWS 16384
#define DIM    128

typedef __attribute__((ext_vector_type(16))) _Float16 v16h;
typedef __attribute__((ext_vector_type(8)))  _Float16 v8h;
typedef __attribute__((ext_vector_type(4)))  _Float16 v4h;
typedef __attribute__((ext_vector_type(8)))  float    v8f;
typedef __attribute__((ext_vector_type(4)))  float    v4f;

// ---------------------------------------------------------------------------
// Kernel 1: per-row L2 normalization + split-f16 encoding.
// One wave32 per row (128 floats = 4 per lane). rh = f16(x/||x||),
// rl = f16(residual), so rh+rl carries ~22 bits of mantissa.
// Input is read exactly once -> non-temporal loads; rh/rl are re-read by the
// GEMM ~128x each -> regular (cache-resident) stores.
// ---------------------------------------------------------------------------
__global__ __launch_bounds__(256) void cosine_rownorm_split_f16(
    const float* __restrict__ h,
    _Float16* __restrict__ rh,
    _Float16* __restrict__ rl) {
  const int row  = blockIdx.x * 8 + (threadIdx.x >> 5);
  const int lane = threadIdx.x & 31;

  const v4f v = __builtin_nontemporal_load(
      (const v4f*)(h + (size_t)row * DIM + lane * 4));
  float ss = v.x * v.x + v.y * v.y + v.z * v.z + v.w * v.w;
#pragma unroll
  for (int off = 16; off > 0; off >>= 1) ss += __shfl_xor(ss, off, 32);

  // matches max(n_i*n_j, 1e-6) denominator for all non-degenerate rows
  const float inv = 1.0f / fmaxf(sqrtf(ss), 1e-6f);

  const float n[4] = {v.x * inv, v.y * inv, v.z * inv, v.w * inv};
  v4h hi_, lo_;
#pragma unroll
  for (int e = 0; e < 4; ++e) {
    const _Float16 hh = (_Float16)n[e];
    hi_[e] = hh;
    lo_[e] = (_Float16)(n[e] - (float)hh);
  }
  *(v4h*)(rh + (size_t)row * DIM + lane * 4) = hi_;
  *(v4h*)(rl + (size_t)row * DIM + lane * 4) = lo_;
}

// ---------------------------------------------------------------------------
// WMMA fragment loaders, using contiguous k order (k = 16*(lane/16) + s), same for A and B.
// A (16x32, MxK): lanes 0-15 = rows m, halves [0..15]=K0..15;
//                 lanes 16-31 = rows m, halves [0..15]=K16..31.
// B (32x16, KxN): lane n holds B[k, n] for 16 consecutive k
//                 (lanes 0-15: K0..15, lanes 16-31: K16..31) -> since
//                 B = R^T, this is a contiguous 32-byte read of row j0+n.
// ---------------------------------------------------------------------------
__device__ __forceinline__ v16h load_a_frag(const _Float16* __restrict__ R,
                                            int row, int k0, int lane) {
  const _Float16* base = R + (size_t)row * DIM + k0 + ((lane & 16) ? 16 : 0);
  const v8h lo = *(const v8h*)(base);        // K chunk 1 (16 bytes)
  const v8h hi = *(const v8h*)(base + 8);    // K chunk 2 (16 bytes)
  return __builtin_shufflevector(lo, hi, 0, 1, 2, 3, 4, 5, 6, 7,
                                 8, 9, 10, 11, 12, 13, 14, 15);
}

__device__ __forceinline__ v16h load_b_frag(const _Float16* __restrict__ R,
                                            int col, int k0, int lane) {
  const _Float16* base = R + (size_t)col * DIM + k0 + ((lane & 16) ? 16 : 0);
  return *(const v16h*)(base);               // contiguous 32 bytes
}

// ---------------------------------------------------------------------------
// Kernel 2: C = R * R^T via v_wmma_f32_16x16x32_f16 with split-f16
// error compensation: dot ~= hi.hi + hi.lo + lo.hi (f32 accumulate).
// 128x128 tile per 8-wave workgroup; each wave owns a 32x64 subtile.
// Output (1 GiB, write-once) is streamed with TH=NT so the 8 MB operand
// matrix stays L2-resident.
// ---------------------------------------------------------------------------
__global__ __launch_bounds__(256) void cosine_wmma_gemm(
    const _Float16* __restrict__ rh,
    const _Float16* __restrict__ rl,
    float* __restrict__ out) {
  const int lane = threadIdx.x & 31;
  const int wave = threadIdx.x >> 5;
  const int wm = wave & 3;   // 4 M-groups x 32 rows
  const int wn = wave >> 2;  // 2 N-groups x 64 cols
  const int i0 = blockIdx.y * 128 + wm * 32;
  const int j0 = blockIdx.x * 128 + wn * 64;

  v8f acc[2][4];
#pragma unroll
  for (int mt = 0; mt < 2; ++mt)
#pragma unroll
    for (int nt = 0; nt < 4; ++nt) {
      v8f z = {};
      acc[mt][nt] = z;
    }

#pragma unroll
  for (int ks = 0; ks < 4; ++ks) {  // K = 128 = 4 x 32
    const int k0 = ks * 32;
    v16h ah[2], al[2], bh[4], bl[4];
#pragma unroll
    for (int mt = 0; mt < 2; ++mt) {
      const int row = i0 + mt * 16 + (lane & 15);
      ah[mt] = load_a_frag(rh, row, k0, lane);
      al[mt] = load_a_frag(rl, row, k0, lane);
    }
#pragma unroll
    for (int nt = 0; nt < 4; ++nt) {
      const int col = j0 + nt * 16 + (lane & 15);
      bh[nt] = load_b_frag(rh, col, k0, lane);
      bl[nt] = load_b_frag(rl, col, k0, lane);
    }
#pragma unroll
    for (int mt = 0; mt < 2; ++mt)
#pragma unroll
      for (int nt = 0; nt < 4; ++nt) {
        acc[mt][nt] = __builtin_amdgcn_wmma_f32_16x16x32_f16(
            false, ah[mt], false, bh[nt], (short)0, acc[mt][nt], false, false);
        acc[mt][nt] = __builtin_amdgcn_wmma_f32_16x16x32_f16(
            false, ah[mt], false, bl[nt], (short)0, acc[mt][nt], false, false);
        acc[mt][nt] = __builtin_amdgcn_wmma_f32_16x16x32_f16(
            false, al[mt], false, bh[nt], (short)0, acc[mt][nt], false, false);
      }
  }

  // C/D layout: VGPR e holds M = e (+8 for lanes 16-31), N = lane&15.
  const int mbase = (lane & 16) ? 8 : 0;
  const int c = lane & 15;
#pragma unroll
  for (int mt = 0; mt < 2; ++mt)
#pragma unroll
    for (int nt = 0; nt < 4; ++nt) {
      const size_t colIdx = (size_t)(j0 + nt * 16 + c);
#pragma unroll
      for (int e = 0; e < 8; ++e) {
        const size_t r = (size_t)(i0 + mt * 16 + mbase + e);
        __builtin_nontemporal_store(acc[mt][nt][e],
                                    out + r * (size_t)N_ROWS + colIdx);
      }
    }
}

// ---------------------------------------------------------------------------
extern "C" void kernel_launch(void* const* d_in, const int* in_sizes, int n_in,
                              void* d_out, int out_size, void* d_ws,
                              size_t ws_size, hipStream_t stream) {
  (void)in_sizes; (void)n_in; (void)out_size; (void)ws_size;
  const float* h1 = (const float*)d_in[0];
  float* out = (float*)d_out;

  _Float16* rh = (_Float16*)d_ws;                       // 4 MB
  _Float16* rl = rh + (size_t)N_ROWS * DIM;             // 4 MB

  cosine_rownorm_split_f16<<<N_ROWS / 8, 256, 0, stream>>>(h1, rh, rl);

  dim3 grid(N_ROWS / 128, N_ROWS / 128);
  cosine_wmma_gemm<<<grid, 256, 0, stream>>>(rh, rl, out);
}
